// FuseBlock_38242388803741
// MI455X (gfx1250) — matrix operand order repaired and checked by reading, not compiled, not run on hardware
//
#include <hip/hip_runtime.h>
#include <hip/hip_bf16.h>
#include <math.h>

// ---------------------------------------------------------------------------
// Types
// ---------------------------------------------------------------------------
typedef _Float16 v16h __attribute__((ext_vector_type(16)));
typedef float    v8f  __attribute__((ext_vector_type(8)));
typedef unsigned int u32x4 __attribute__((ext_vector_type(4)));
typedef int          i32x4v __attribute__((ext_vector_type(4)));
typedef int          i32x8v __attribute__((ext_vector_type(8)));

union FragH { v16h h; unsigned u[8]; };

#define NB_   4
#define C_    64
#define H_    256
#define W_    256
#define S_    65536      // H_*W_
#define KB_   129        // rfft bins
#define SF_   33024      // H_*KB_
#define HEADS_ 8

// ---------------------------------------------------------------------------
// LDS offset helper (generic ptr low 32 bits == LDS byte offset per aperture)
// ---------------------------------------------------------------------------
__device__ inline unsigned lds_off(const void* p) {
    return (unsigned)(unsigned long long)(p);
}

// ---------------------------------------------------------------------------
// TDM 2D copy: rows x rowlen f32 elements, row stride = rowstride (elements)
// ---------------------------------------------------------------------------
#if __has_builtin(__builtin_amdgcn_tensor_load_to_lds)
#define HAVE_TDM 1
__device__ inline void tdm_load_f32_2d(const float* gsrc, unsigned ldsByteOff,
                                       int rows, int rowlen, int rowstride) {
    unsigned long long ga = (unsigned long long)(const void*)gsrc;
    u32x4 g0;
    g0[0] = 1u;                                   // count=1, user mode, no gather
    g0[1] = ldsByteOff;                           // lds_addr
    g0[2] = (unsigned)(ga & 0xFFFFFFFFull);       // global_addr[31:0]
    g0[3] = (unsigned)((ga >> 32) & 0x01FFFFFFull) | (2u << 30); // addr[56:32] | type=2
    i32x8v g1;
    g1[0] = (int)(2u << 16);                                          // data_size=4B
    g1[1] = (int)(((unsigned)rowlen & 0xFFFFu) << 16);                // tensor_dim0 lo
    g1[2] = (int)((((unsigned)rowlen >> 16) & 0xFFFFu) |
                  (((unsigned)rows & 0xFFFFu) << 16));                // dim0 hi | dim1 lo
    g1[3] = (int)((((unsigned)rows >> 16) & 0xFFFFu) |
                  (((unsigned)rowlen & 0xFFFFu) << 16));              // dim1 hi | tile_dim0
    g1[4] = (int)((unsigned)rows & 0xFFFFu);                          // tile_dim1 (tile_dim2=0)
    g1[5] = (int)(unsigned)rowstride;                                 // dim0_stride lo32
    g1[6] = 0; g1[7] = 0;
    i32x4v z4 = {0, 0, 0, 0};
#if defined(__clang_major__) && (__clang_major__ >= 23)
    i32x8v z8 = {0, 0, 0, 0, 0, 0, 0, 0};
    __builtin_amdgcn_tensor_load_to_lds(g0, g1, z4, z4, z8, 0);
#else
    __builtin_amdgcn_tensor_load_to_lds(g0, g1, z4, z4, 0);
#endif
}
#endif

// ---------------------------------------------------------------------------
// Generic WMMA implicit-GEMM conv: Y[b,co,sp] = act( W[co,:] . im2col(X) + bias ) (+resid)
//  - KSZ==1 : 1x1 conv, K=CIN, flat spatial
//  - KSZ==3 : 3x3 conv pad=1, K=CIN*9, 16-pixel tiles along x (W % 16 == 0)
// Block = 128 threads (4 waves); wave w owns Cout rows [blockIdx.y*64 + w*16, +16),
// 16 pixel columns. K staged in 64-wide chunks: TDM f32 stage -> f16 LDS.
// ---------------------------------------------------------------------------
template <int COUT, int CIN, int KSZ, bool RELU, bool HASBIAS, bool RESID>
__global__ __launch_bounds__(128) void wmma_conv(
        const float* __restrict__ X, const float* __restrict__ Wt,
        const float* __restrict__ Bs, const float* __restrict__ R,
        float* __restrict__ Y, int NB, int H, int W) {
    constexpr int K  = CIN * KSZ * KSZ;
    constexpr int KC = 64;
    const int S = H * W;

    __shared__ float    sWst[64 * KC];   // f32 weight staging (16 KB)
    __shared__ _Float16 sWh [64 * KC];   // f16 weights (8 KB)
    __shared__ _Float16 sB  [16 * KC];   // f16 B tile [n][kk] (2 KB)

    const int tid  = threadIdx.x;
    const int wave = tid >> 5;
    const int lane = tid & 31;
    const int mBlk = blockIdx.y * 64;

    const long g0p = (long)blockIdx.x * 16;   // first flat pixel over NB*S
    const int  b   = (int)(g0p / S);
    const int  sp0 = (int)(g0p % S);
    const int  y0  = sp0 / W;
    const int  x0  = sp0 % W;

    if (KSZ == 1) __builtin_prefetch(&X[((long)b * CIN) * (long)S + sp0], 0, 1);

    v8f acc = {0.f, 0.f, 0.f, 0.f, 0.f, 0.f, 0.f, 0.f};

    for (int kc = 0; kc < K; kc += KC) {
        // ---- stage weight chunk rows [mBlk, mBlk+64) cols [kc, kc+64) ----
#ifdef HAVE_TDM
        if (tid < 32) {
            tdm_load_f32_2d(Wt + (long)mBlk * K + kc, lds_off(sWst), 64, KC, K);
            __builtin_amdgcn_s_wait_tensorcnt(0);
        }
#else
        for (int i = tid; i < 64 * KC; i += 128)
            sWst[i] = Wt[(long)(mBlk + (i >> 6)) * K + kc + (i & 63)];
#endif
        __syncthreads();
        // convert weights to f16
        for (int i = tid; i < 64 * KC; i += 128)
            sWh[i] = (_Float16)sWst[i];
        // ---- load B tile: 16 pixels x 64 k ----
        for (int i = tid; i < 16 * KC; i += 128) {
            const int n  = i >> 6;
            const int kk = i & 63;
            const int kg = kc + kk;
            float v = 0.f;
            if (KSZ == 1) {
                v = X[((long)b * CIN + kg) * (long)S + (sp0 + n)];
            } else {
                const int ci  = kg / 9;
                const int tap = kg % 9;
                const int yy  = y0 + tap / 3 - 1;
                const int xx  = x0 + n + (tap % 3) - 1;
                if (yy >= 0 && yy < H && xx >= 0 && xx < W)
                    v = X[((long)b * CIN + ci) * (long)S + (long)yy * W + xx];
            }
            sB[i] = (_Float16)v;
        }
        __syncthreads();

        // ---- fragments + WMMA (ISA 16-bit A 16x32 / B 32x16 layouts) ----
        const unsigned* wA32 = (const unsigned*)sWh;
        const unsigned* bB32 = (const unsigned*)sB;
        const int  mrow = wave * 16 + (lane & 15);
        const bool hi   = lane >= 16;
        const int  ncol = lane & 15;
#pragma unroll
        for (int s = 0; s < 2; ++s) {
            FragH fa, fb;
            const unsigned abase = (unsigned)(mrow * KC + s * 32 + (hi ? 8 : 0)) >> 1;
#pragma unroll
            for (int j = 0; j < 4; ++j) {
                fa.u[j]     = wA32[abase + j];       // K halves [base, base+8)
                fa.u[4 + j] = wA32[abase + 8 + j];   // K halves [base+16, base+24)
            }
            const unsigned bbase = (unsigned)(ncol * KC + s * 32 + (hi ? 8 : 0)) >> 1;
#pragma unroll
            for (int j = 0; j < 4; ++j) { fb.u[j] = bB32[bbase + j]; fb.u[4 + j] = bB32[bbase + 8 + j]; }
            acc = __builtin_amdgcn_wmma_f32_16x16x32_f16(
                false, fa.h, false, fb.h, (short)0, acc, false, false);
        }
        __syncthreads();
    }

    // ---- writeback (C/D layout: lane<16 -> M=r, lane>=16 -> M=8+r, N=lane&15) ----
    const int n    = lane & 15;
    const int sp   = sp0 + n;
    const int mtop = mBlk + wave * 16 + ((lane >= 16) ? 8 : 0);
#pragma unroll
    for (int r = 0; r < 8; ++r) {
        const int co = mtop + r;
        float v = acc[r];
        if (HASBIAS) v += Bs[co];
        if (RELU)    v = fmaxf(v, 0.f);
        const long oi = ((long)b * COUT + co) * (long)S + sp;
        if (RESID)   v += R[oi];
        Y[oi] = v;
    }
}

// ---------------------------------------------------------------------------
// Stockham radix-2 complex FFT, N=256, 128 threads. Result ends in buf0.
// sgn=+1 forward, -1 inverse (unscaled).
// ---------------------------------------------------------------------------
__device__ inline void fft256_lds(float* r0, float* i0, float* r1, float* i1,
                                  int t, float sgn) {
    float* sR = r0; float* sI = i0; float* dR = r1; float* dI = i1;
#pragma unroll
    for (int st = 0; st < 8; ++st) {
        const int L = 1 << st;
        const int k = t & (L - 1);
        const float ar = sR[t],        ai = sI[t];
        const float br = sR[t + 128],  bi = sI[t + 128];
        const float ang = sgn * (-3.14159265358979323846f) * (float)k / (float)L;
        const float wr = __cosf(ang), wi = __sinf(ang);
        const float tr = wr * br - wi * bi;
        const float ti = wr * bi + wi * br;
        const int j = ((t - k) << 1) + k;
        dR[j]     = ar + tr; dI[j]     = ai + ti;
        dR[j + L] = ar - tr; dI[j + L] = ai - ti;
        __syncthreads();
        float* tp;
        tp = sR; sR = dR; dR = tp;
        tp = sI; sI = dI; dI = tp;
    }
}

// rows rfft: X[b,c,y,:] (real 256) -> F real plane c, imag plane 64+c, bins 0..128
__global__ __launch_bounds__(128) void rfft_rows(const float* __restrict__ X,
                                                 float* __restrict__ F) {
    __shared__ float r0[256], i0[256], r1[256], i1[256];
    const int t  = threadIdx.x;
    const int id = blockIdx.x;
    const int y  = id % H_;
    const int c  = (id / H_) % C_;
    const int b  = id / (H_ * C_);
    const float* row = X + ((long)(b * C_ + c)) * (long)S_ + (long)y * W_;
    r0[t] = row[t]; r0[t + 128] = row[t + 128];
    i0[t] = 0.f;    i0[t + 128] = 0.f;
    __syncthreads();
    fft256_lds(r0, i0, r1, i1, t, 1.f);
    const long pr = ((long)(b * 128 + c) * H_ + y) * (long)KB_;
    const long pi = ((long)(b * 128 + 64 + c) * H_ + y) * (long)KB_;
    F[pr + t] = r0[t]; F[pi + t] = i0[t];
    if (t == 0) { F[pr + 128] = r0[128]; F[pi + 128] = i0[128]; }
}

// column complex FFT over y (in-place), per (b, c, kbin)
__global__ __launch_bounds__(128) void colfft(float* __restrict__ F, float sgn) {
    __shared__ float r0[256], i0[256], r1[256], i1[256];
    const int t  = threadIdx.x;
    const int id = blockIdx.x;
    const int kb = id % KB_;
    const int c  = (id / KB_) % C_;
    const int b  = id / (KB_ * C_);
    const long pr = ((long)(b * 128 + c) * H_) * (long)KB_ + kb;
    const long pi = ((long)(b * 128 + 64 + c) * H_) * (long)KB_ + kb;
    r0[t] = F[pr + (long)t * KB_];        r0[t + 128] = F[pr + (long)(t + 128) * KB_];
    i0[t] = F[pi + (long)t * KB_];        i0[t + 128] = F[pi + (long)(t + 128) * KB_];
    __syncthreads();
    fft256_lds(r0, i0, r1, i1, t, sgn);
    F[pr + (long)t * KB_] = r0[t];        F[pr + (long)(t + 128) * KB_] = r0[t + 128];
    F[pi + (long)t * KB_] = i0[t];        F[pi + (long)(t + 128) * KB_] = i0[t + 128];
}

// rows irfft: hermitian-extend bins 0..128, inverse FFT, real out * 1/(H*W)
__global__ __launch_bounds__(128) void irfft_rows(const float* __restrict__ F,
                                                  float* __restrict__ Y) {
    __shared__ float r0[256], i0[256], r1[256], i1[256];
    const int t  = threadIdx.x;
    const int id = blockIdx.x;
    const int y  = id % H_;
    const int c  = (id / H_) % C_;
    const int b  = id / (H_ * C_);
    const long pr = ((long)(b * 128 + c) * H_ + y) * (long)KB_;
    const long pi = ((long)(b * 128 + 64 + c) * H_ + y) * (long)KB_;
    r0[t] = F[pr + t]; i0[t] = F[pi + t];
    if (t == 0) { r0[128] = F[pr + 128]; i0[128] = F[pi + 128]; }
    __syncthreads();
    if (t >= 1) {  // bins 129..255 = conj(256-k)
        r0[128 + t] =  r0[128 - t];
        i0[128 + t] = -i0[128 - t];
    }
    __syncthreads();
    fft256_lds(r0, i0, r1, i1, t, -1.f);
    const float sc = 1.f / 65536.f;
    float* row = Y + ((long)(b * C_ + c)) * (long)S_ + (long)y * W_;
    row[t] = r0[t] * sc; row[t + 128] = r0[t + 128] * sc;
}

// ---------------------------------------------------------------------------
// Depthwise 3x3 pad=1 (optional channel-concat input, optional bias)
// ---------------------------------------------------------------------------
__global__ void dw3x3(const float* __restrict__ X, const float* __restrict__ X2,
                      int Csplit, const float* __restrict__ Wd,
                      const float* __restrict__ Bs, float* __restrict__ Y,
                      int NB, int Cn, int H, int W) {
    const long total = (long)NB * Cn * H * W;
    for (long idx = (long)blockIdx.x * blockDim.x + threadIdx.x; idx < total;
         idx += (long)gridDim.x * blockDim.x) {
        const int x = (int)(idx % W);
        const int y = (int)((idx / W) % H);
        const int c = (int)((idx / ((long)W * H)) % Cn);
        const int b = (int)(idx / ((long)W * H * Cn));
        const float* src;
        long base;
        if (X2 != nullptr && c >= Csplit) {
            src  = X2;
            base = ((long)b * (Cn - Csplit) + (c - Csplit)) * (long)(H * W);
        } else {
            src  = X;
            base = ((long)b * Csplit + c) * (long)(H * W);
        }
        float acc = 0.f;
#pragma unroll
        for (int tap = 0; tap < 9; ++tap) {
            const int yy = y + tap / 3 - 1;
            const int xx = x + tap % 3 - 1;
            if (yy >= 0 && yy < H && xx >= 0 && xx < W)
                acc += Wd[c * 9 + tap] * src[base + (long)yy * W + xx];
        }
        if (Bs != nullptr) acc += Bs[c];
        Y[idx] = acc;
    }
}

// ---------------------------------------------------------------------------
// L2 norms per (b, channel) over S_ elements; cstride = channels per batch
// ---------------------------------------------------------------------------
__global__ __launch_bounds__(256) void l2norms(const float* __restrict__ X,
                                               int cstride, float* __restrict__ out) {
    __shared__ float sh[256];
    const int bc = blockIdx.x;           // b*64 + c
    const int b  = bc / 64, c = bc % 64;
    const long base = ((long)b * cstride + c) * (long)S_;
    float s = 0.f;
    for (int i = threadIdx.x; i < S_; i += 256) {
        const float v = X[base + i];
        s += v * v;
    }
    sh[threadIdx.x] = s;
    __syncthreads();
    for (int off = 128; off > 0; off >>= 1) {
        if (threadIdx.x < off) sh[threadIdx.x] += sh[threadIdx.x + off];
        __syncthreads();
    }
    if (threadIdx.x == 0) out[bc] = fmaxf(sqrtf(sh[0]), 1e-12f);
}

__global__ void zerok(float* __restrict__ p, int n) {
    const int i = blockIdx.x * blockDim.x + threadIdx.x;
    if (i < n) p[i] = 0.f;
}

// attn_raw[b,h,c,d] += sum_n q[b,h*8+c,n] * k[b,h*8+d,n] ; chunked over n
__global__ __launch_bounds__(256) void att_acc(const float* __restrict__ Q,
                                               const float* __restrict__ KVb,
                                               float* __restrict__ raw) {
    __shared__ float sh[64];
    const int chunk = blockIdx.x & 63;
    const int bh    = blockIdx.x >> 6;
    const int b = bh / HEADS_, h = bh % HEADS_;
    const int t = threadIdx.x;
    float a[64];
#pragma unroll
    for (int i = 0; i < 64; ++i) a[i] = 0.f;
    const int nbase = chunk * 1024;
#pragma unroll
    for (int it = 0; it < 4; ++it) {
        const int n = nbase + t + it * 256;
        float qv[8], kv[8];
#pragma unroll
        for (int i = 0; i < 8; ++i) {
            qv[i] = Q  [((long)(b * 64  + h * 8 + i)) * (long)S_ + n];
            kv[i] = KVb[((long)(b * 128 + h * 8 + i)) * (long)S_ + n];
        }
#pragma unroll
        for (int c = 0; c < 8; ++c)
#pragma unroll
            for (int d = 0; d < 8; ++d) a[c * 8 + d] += qv[c] * kv[d];
    }
    if (t < 64) sh[t] = 0.f;
    __syncthreads();
#pragma unroll
    for (int i = 0; i < 64; ++i) atomicAdd(&sh[i], a[i]);
    __syncthreads();
    if (t < 64) atomicAdd(&raw[(long)bh * 64 + t], sh[t]);
}

// softmax over d of raw/(nq*nk)*temp[h]
__global__ __launch_bounds__(64) void att_softmax(const float* __restrict__ raw,
                                                  const float* __restrict__ NQ,
                                                  const float* __restrict__ NK,
                                                  const float* __restrict__ temp,
                                                  float* __restrict__ P) {
    __shared__ float sv[64];
    __shared__ float red[8];
    const int bh = blockIdx.x;
    const int b = bh / HEADS_, h = bh % HEADS_;
    const int t = threadIdx.x, c = t >> 3, d = t & 7;
    const float nq = NQ[b * 64 + h * 8 + c];
    const float nk = NK[b * 64 + h * 8 + d];
    const float v  = raw[(long)bh * 64 + t] / (nq * nk) * temp[h];
    sv[t] = v;
    __syncthreads();
    if (d == 0) {
        float m = sv[c * 8];
        for (int j = 1; j < 8; ++j) m = fmaxf(m, sv[c * 8 + j]);
        red[c] = m;
    }
    __syncthreads();
    const float e = __expf(v - red[c]);
    sv[t] = e;
    __syncthreads();
    if (d == 0) {
        float s = 0.f;
        for (int j = 0; j < 8; ++j) s += sv[c * 8 + j];
        red[c] = s;
    }
    __syncthreads();
    P[(long)bh * 64 + t] = e / red[c];
}

// out[b, h*8+c, n] = sum_d P[b,h,c,d] * v[b, h*8+d, n]   (v = KV channels 64..127)
__global__ void att_apply(const float* __restrict__ P, const float* __restrict__ KVb,
                          float* __restrict__ O) {
    const long total = (long)NB_ * 64 * S_;
    for (long idx = (long)blockIdx.x * blockDim.x + threadIdx.x; idx < total;
         idx += (long)gridDim.x * blockDim.x) {
        const int n  = (int)(idx % S_);
        const int cg = (int)((idx / S_) % 64);
        const int b  = (int)(idx / ((long)S_ * 64));
        const int h = cg >> 3, c = cg & 7;
        const float* ap = P + ((long)(b * HEADS_ + h)) * 64 + c * 8;
        float s = 0.f;
#pragma unroll
        for (int d = 0; d < 8; ++d)
            s += ap[d] * KVb[((long)(b * 128 + 64 + h * 8 + d)) * (long)S_ + n];
        O[idx] = s;
    }
}

// res = nan_to_num( fre*sig(g[0:64]) + spa*sig(g[64:128]) )
__global__ void fuse_gate(const float* __restrict__ FRE, const float* __restrict__ SPA,
                          const float* __restrict__ G, float* __restrict__ O) {
    const long total = (long)NB_ * 64 * S_;
    for (long idx = (long)blockIdx.x * blockDim.x + threadIdx.x; idx < total;
         idx += (long)gridDim.x * blockDim.x) {
        const int n = (int)(idx % S_);
        const int c = (int)((idx / S_) % 64);
        const int b = (int)(idx / ((long)S_ * 64));
        const float gf = G[((long)b * 128 + c) * (long)S_ + n];
        const float gs = G[((long)b * 128 + 64 + c) * (long)S_ + n];
        const float af = 1.f / (1.f + __expf(-gf));
        const float as = 1.f / (1.f + __expf(-gs));
        float r = FRE[idx] * af + SPA[idx] * as;
        if (!isfinite(r)) r = 1e-5f;
        O[idx] = r;
    }
}

// ---------------------------------------------------------------------------
// Host side
// ---------------------------------------------------------------------------
struct AttW {
    const float *kv_w, *kv_dw, *q_w, *q_dw, *proj_w, *temp;
};
struct AttB {
    float *KVPRE, *KV, *T1, *QB, *NQ, *NK, *RAW, *PRB;
};

static void run_attention(const float* xq, const float* ykv, const float* resid,
                          float* outbuf, const AttW& w, const AttB& s,
                          hipStream_t st) {
    const dim3 blk(128);
    wmma_conv<128, 64, 1, false, false, false><<<dim3(16384, 2), blk, 0, st>>>(
        ykv, w.kv_w, nullptr, nullptr, s.KVPRE, NB_, H_, W_);
    dw3x3<<<131072, 256, 0, st>>>(s.KVPRE, nullptr, 128, w.kv_dw, nullptr, s.KV,
                                  NB_, 128, H_, W_);
    wmma_conv<64, 64, 1, false, false, false><<<dim3(16384, 1), blk, 0, st>>>(
        xq, w.q_w, nullptr, nullptr, s.T1, NB_, H_, W_);
    dw3x3<<<65536, 256, 0, st>>>(s.T1, nullptr, 64, w.q_dw, nullptr, s.QB,
                                 NB_, 64, H_, W_);
    l2norms<<<256, 256, 0, st>>>(s.QB, 64, s.NQ);
    l2norms<<<256, 256, 0, st>>>(s.KV, 128, s.NK);
    zerok<<<8, 256, 0, st>>>(s.RAW, 2048);
    att_acc<<<NB_ * HEADS_ * 64, 256, 0, st>>>(s.QB, s.KV, s.RAW);
    att_softmax<<<NB_ * HEADS_, 64, 0, st>>>(s.RAW, s.NQ, s.NK, w.temp, s.PRB);
    att_apply<<<65536, 256, 0, st>>>(s.PRB, s.KV, s.T1);
    wmma_conv<64, 64, 1, false, false, true><<<dim3(16384, 1), blk, 0, st>>>(
        s.T1, w.proj_w, nullptr, resid, outbuf, NB_, H_, W_);
}

extern "C" void kernel_launch(void* const* d_in, const int* in_sizes, int n_in,
                              void* d_out, int out_size, void* d_ws, size_t ws_size,
                              hipStream_t stream) {
    (void)in_sizes; (void)n_in; (void)out_size; (void)ws_size;
    const float* x0        = (const float*)d_in[0];
    const float* x1        = (const float*)d_in[1];
    const float* rl_w1     = (const float*)d_in[2];
    const float* rl_b1     = (const float*)d_in[3];
    const float* rl_w2     = (const float*)d_in[4];
    const float* rl_b2     = (const float*)d_in[5];
    const float* rg_w1     = (const float*)d_in[6];
    const float* rg_w2     = (const float*)d_in[7];
    const float* att_temp  = (const float*)d_in[8];
    const float* att_kv_w  = (const float*)d_in[9];
    const float* att_kv_dw = (const float*)d_in[10];
    const float* att_q_w   = (const float*)d_in[11];
    const float* att_q_dw  = (const float*)d_in[12];
    const float* att_proj  = (const float*)d_in[13];
    const float* f1_dw     = (const float*)d_in[14];
    const float* f1_dwb    = (const float*)d_in[15];
    const float* f1_pw     = (const float*)d_in[16];
    const float* f1_pwb    = (const float*)d_in[17];
    const float* f2_dw     = (const float*)d_in[18];
    const float* f2_dwb    = (const float*)d_in[19];
    const float* f2_pw     = (const float*)d_in[20];
    const float* f2_pwb    = (const float*)d_in[21];
    float* out = (float*)d_out;

    // workspace partition (floats)
    const size_t S64  = (size_t)NB_ * 64 * S_;          // 16,777,216
    const size_t SFq  = (size_t)NB_ * 128 * SF_;        // 16,908,288 (freq planes)
    float* ws   = (float*)d_ws;
    float* T1   = ws;                 // B64
    float* FRE  = ws + 1 * S64;       // B64
    float* SPA  = ws + 2 * S64;       // B64
    float* QB   = ws + 3 * S64;       // B64
    float* FRE2 = ws + 4 * S64;       // B64
    float* SPA2 = ws + 5 * S64;       // B64
    float* KVPRE= ws + 6 * S64;       // B128 (aliases F0; freq dead by stage 3)
    float* KV   = ws + 8 * S64;       // B128
    float* F0   = KVPRE;              // SFq
    float* F1   = F0 + SFq;           // SFq (tail spills into KV head; dead before KV use)
    float* SMALL= ws + 10 * S64;
    float* NQ   = SMALL;              // 256
    float* NK   = SMALL + 256;        // 256
    float* RAW  = SMALL + 512;        // 2048
    float* PRB  = SMALL + 2560;       // 2048

    const dim3 blk(128);

    // ---- Stage 1: ResBlock_L : fre = relu(conv(relu(conv(x0)))) + x0 ----
    wmma_conv<64, 64, 3, true, true, false><<<dim3(16384, 1), blk, 0, stream>>>(
        x0, rl_w1, rl_b1, nullptr, T1, NB_, H_, W_);
    wmma_conv<64, 64, 3, true, true, true><<<dim3(16384, 1), blk, 0, stream>>>(
        T1, rl_w2, rl_b2, x0, FRE, NB_, H_, W_);

    // ---- Stage 2: ResBlock_G (FFT domain) -> spa ----
    rfft_rows<<<NB_ * C_ * H_, 128, 0, stream>>>(x1, F0);
    colfft<<<NB_ * C_ * KB_, 128, 0, stream>>>(F0, 1.f);
    wmma_conv<128, 128, 1, true, false, false><<<dim3(8256, 2), blk, 0, stream>>>(
        F0, rg_w1, nullptr, nullptr, F1, NB_, H_, KB_);
    wmma_conv<128, 128, 1, false, false, false><<<dim3(8256, 2), blk, 0, stream>>>(
        F1, rg_w2, nullptr, nullptr, F0, NB_, H_, KB_);
    colfft<<<NB_ * C_ * KB_, 128, 0, stream>>>(F0, -1.f);
    irfft_rows<<<NB_ * C_ * H_, 128, 0, stream>>>(F0, SPA);

    // ---- Stages 3/4: shared attention, twice ----
    AttW aw{att_kv_w, att_kv_dw, att_q_w, att_q_dw, att_proj, att_temp};
    AttB ab{KVPRE, KV, T1, QB, NQ, NK, RAW, PRB};
    run_attention(FRE, SPA, FRE, FRE2, aw, ab, stream);   // fre = att(fre, spa) + fre
    run_attention(SPA, FRE2, SPA, SPA2, aw, ab, stream);  // spa = att(spa, fre) + spa

    // ---- Stage 5: fuse gate ----
    dw3x3<<<131072, 256, 0, stream>>>(FRE2, SPA2, 64, f1_dw, f1_dwb, KVPRE,
                                      NB_, 128, H_, W_);
    wmma_conv<64, 128, 1, false, true, false><<<dim3(16384, 1), blk, 0, stream>>>(
        KVPRE, f1_pw, f1_pwb, nullptr, T1, NB_, H_, W_);
    dw3x3<<<65536, 256, 0, stream>>>(T1, nullptr, 64, f2_dw, f2_dwb, QB,
                                     NB_, 64, H_, W_);
    wmma_conv<128, 64, 1, false, true, false><<<dim3(16384, 2), blk, 0, stream>>>(
        QB, f2_pw, f2_pwb, nullptr, KV, NB_, H_, W_);
    fuse_gate<<<65536, 256, 0, stream>>>(FRE2, SPA2, KV, out);
}
